// MeshGraphNet_34643206210081
// MI455X (gfx1250) — matrix operand order repaired and checked by reading, not compiled, not run on hardware
//
#include <hip/hip_runtime.h>

#define NNODES 50000
#define NEDGES 800000
#define LAT 128
#define NSTEPS 15

typedef __attribute__((ext_vector_type(16))) __bf16 bf16x16;
typedef __attribute__((ext_vector_type(8)))  float  floatx8;

union Frag { bf16x16 v; uint4 q[2]; };
union Acc  { floatx8 v; float f[8]; };

__device__ __forceinline__ unsigned short f2bf(float f) {
    unsigned int u = __float_as_uint(f);
    u += 0x7FFFu + ((u >> 16) & 1u);           // round-to-nearest-even
    return (unsigned short)(u >> 16);
}
__device__ __forceinline__ float bf2f(unsigned short h) {
    return __uint_as_float(((unsigned int)h) << 16);
}

__device__ __forceinline__ floatx8 wmma_bf16(bf16x16 a, bf16x16 b, floatx8 c) {
    // 8 args: (neg_a, A, neg_b, B, c_mod, C, reuse_a, reuse_b)
    return __builtin_amdgcn_wmma_f32_16x16x32_bf16(false, a, false, b,
                                                   (short)0, c, false, false);
}

// ---------------------------------------------------------------------------
// Weight prep: f32 [S][K][N] -> bf16 [S][N][K] (B-matrix friendly layout)
// ---------------------------------------------------------------------------
__global__ void transpose_bf16_kernel(const float* __restrict__ in,
                                      unsigned short* __restrict__ out,
                                      int S, int K, int N)
{
    long total = (long)S * K * N;
    for (long i = (long)blockIdx.x * blockDim.x + threadIdx.x; i < total;
         i += (long)gridDim.x * blockDim.x) {
        long s = i / ((long)K * N);
        long rem = i - s * (long)K * N;
        long k = rem / N;
        long n = rem - k * N;
        out[(s * N + n) * K + k] = f2bf(in[i]);
    }
}

// ---------------------------------------------------------------------------
// Node encoder: 8 nodes / 128-thread block. h = MLP2(norm(x))
// ---------------------------------------------------------------------------
__global__ __launch_bounds__(128) void node_enc_kernel(
    const float* __restrict__ node_in, const float* __restrict__ mean,
    const float* __restrict__ stdv,
    const float* __restrict__ W1, const float* __restrict__ b1,
    const float* __restrict__ W2, const float* __restrict__ b2,
    float* __restrict__ h_f32, unsigned short* __restrict__ h_bf)
{
    __shared__ float x_sm[8][11];
    __shared__ float hid_sm[8][129];
    const long nbase = (long)blockIdx.x * 8;
    const int tid = threadIdx.x;
    if (tid < 88) {
        int row = tid / 11, col = tid % 11;
        x_sm[row][col] = (node_in[(nbase + row) * 11 + col] - mean[col]) / stdv[col];
    }
    __syncthreads();
    float acc[8];
    #pragma unroll
    for (int e = 0; e < 8; ++e) acc[e] = b1[tid];
    for (int k = 0; k < 11; ++k) {
        float w = W1[k * LAT + tid];
        #pragma unroll
        for (int e = 0; e < 8; ++e) acc[e] += x_sm[e][k] * w;
    }
    #pragma unroll
    for (int e = 0; e < 8; ++e) hid_sm[e][tid] = fmaxf(acc[e], 0.f);
    __syncthreads();
    #pragma unroll
    for (int e = 0; e < 8; ++e) acc[e] = b2[tid];
    for (int k = 0; k < LAT; ++k) {
        float w = W2[k * LAT + tid];
        #pragma unroll
        for (int e = 0; e < 8; ++e) acc[e] += hid_sm[e][k] * w;
    }
    #pragma unroll
    for (int e = 0; e < 8; ++e) {
        long idx = (nbase + e) * LAT + tid;
        h_f32[idx] = acc[e];
        h_bf[idx]  = f2bf(acc[e]);
    }
}

// ---------------------------------------------------------------------------
// Edge encoder: 16 edges / 128-thread block. e = MLP2(norm(x)) (bf16 out)
// ---------------------------------------------------------------------------
__global__ __launch_bounds__(128) void edge_enc_kernel(
    const float* __restrict__ edge_in, const float* __restrict__ mean,
    const float* __restrict__ stdv,
    const float* __restrict__ W1, const float* __restrict__ b1,
    const float* __restrict__ W2, const float* __restrict__ b2,
    unsigned short* __restrict__ e_bf)
{
    __shared__ float x_sm[16][3];
    __shared__ float hid_sm[16][129];
    const long ebase = (long)blockIdx.x * 16;
    const int tid = threadIdx.x;
    if (tid < 48) {
        int row = tid / 3, col = tid % 3;
        x_sm[row][col] = (edge_in[(ebase + row) * 3 + col] - mean[col]) / stdv[col];
    }
    __syncthreads();
    float acc[16];
    #pragma unroll
    for (int e = 0; e < 16; ++e) acc[e] = b1[tid];
    for (int k = 0; k < 3; ++k) {
        float w = W1[k * LAT + tid];
        #pragma unroll
        for (int e = 0; e < 16; ++e) acc[e] += x_sm[e][k] * w;
    }
    #pragma unroll
    for (int e = 0; e < 16; ++e) hid_sm[e][tid] = fmaxf(acc[e], 0.f);
    __syncthreads();
    #pragma unroll
    for (int e = 0; e < 16; ++e) acc[e] = b2[tid];
    for (int k = 0; k < LAT; ++k) {
        float w = W2[k * LAT + tid];
        #pragma unroll
        for (int e = 0; e < 16; ++e) acc[e] += hid_sm[e][k] * w;
    }
    #pragma unroll
    for (int e = 0; e < 16; ++e)
        e_bf[(ebase + e) * LAT + tid] = f2bf(acc[e]);
}

// ---------------------------------------------------------------------------
// Edge message kernel (WMMA): 64 edges per 256-thread block (8 wave32),
// 2x2 output tiles per wave (4 WMMAs per K-chunk per wave).
//   m = relu([e|h_s|h_r] @ W1 + b1) @ W2 + b2 ; e += m ; agg[recv] += m
// Dynamic LDS: A 64x392 bf16 + hidden 64x136 bf16 = 67584 B.
// ---------------------------------------------------------------------------
#define EMSG_LDS_BYTES ((64 * 392 + 64 * 136) * 2)

__global__ __launch_bounds__(256) void edge_msg_kernel(
    unsigned short* __restrict__ e_bf,          // [E][128] bf16, updated in place
    const unsigned short* __restrict__ h_bf,    // [N][128] bf16
    const int* __restrict__ senders,
    const int* __restrict__ receivers,
    const unsigned short* __restrict__ W1t,     // [128][384] bf16 (N-major)
    const float* __restrict__ b1,
    const unsigned short* __restrict__ W2t,     // [128][128] bf16 (N-major)
    const float* __restrict__ b2,
    float* __restrict__ agg)                    // [N][128] f32 (atomic)
{
    extern __shared__ __align__(16) unsigned short dsm[];
    unsigned short* a_sm   = dsm;              // 64 x 392 (384 + pad)
    unsigned short* hid_sm = dsm + 64 * 392;   // 64 x 136 (128 + pad)
    __shared__ int sidx[64];
    __shared__ int ridx[64];

    const int tid = threadIdx.x;
    const long ebase = (long)blockIdx.x * 64;

    if (tid < 64) { sidx[tid] = senders[ebase + tid]; ridx[tid] = receivers[ebase + tid]; }
    __syncthreads();

    // Build A = [e | h_s | h_r] rows in LDS, 16B chunks (64 rows x 48 chunks)
    for (int c = tid; c < 64 * 48; c += 256) {
        int row = c / 48, j = c % 48;
        const uint4* src;
        if (j < 16)      src = ((const uint4*)(e_bf + (ebase + row) * LAT)) + j;
        else if (j < 32) src = ((const uint4*)(h_bf + (long)sidx[row] * LAT)) + (j - 16);
        else             src = ((const uint4*)(h_bf + (long)ridx[row] * LAT)) + (j - 32);
        *(uint4*)(a_sm + row * 392 + j * 8) = *src;
    }
    __syncthreads();

    const int wave = tid >> 5, lane = tid & 31;
    const int mt0 = (wave >> 2) * 2;     // M-tiles mt0, mt0+1 (of 4)
    const int nt0 = (wave & 3) * 2;      // N-tiles nt0, nt0+1 (of 8)
    const int lr  = lane & 15;
    const int hi  = lane >> 4;           // lane half selects K sub-block

    Acc acc[2][2];
    #pragma unroll
    for (int mi = 0; mi < 2; ++mi)
        #pragma unroll
        for (int ni = 0; ni < 2; ++ni)
            #pragma unroll
            for (int i = 0; i < 8; ++i) acc[mi][ni].f[i] = 0.f;

    // GEMM1: [64x384] @ [384x128], K-chunks of 32
    {
        const unsigned short* arow0 = a_sm + (mt0 * 16 + lr) * 392 + hi * 16;
        const unsigned short* arow1 = arow0 + 16 * 392;
        const unsigned short* brow0 = W1t + (nt0 * 16 + lr) * 384 + hi * 16;
        const unsigned short* brow1 = brow0 + 16 * 384;
        #pragma unroll
        for (int kc = 0; kc < 12; ++kc) {
            Frag A0, A1, B0, B1;
            A0.q[0] = *(const uint4*)(arow0 + kc * 32);
            A0.q[1] = *(const uint4*)(arow0 + kc * 32 + 8);
            A1.q[0] = *(const uint4*)(arow1 + kc * 32);
            A1.q[1] = *(const uint4*)(arow1 + kc * 32 + 8);
            B0.q[0] = *(const uint4*)(brow0 + kc * 32);
            B0.q[1] = *(const uint4*)(brow0 + kc * 32 + 8);
            B1.q[0] = *(const uint4*)(brow1 + kc * 32);
            B1.q[1] = *(const uint4*)(brow1 + kc * 32 + 8);
            acc[0][0].v = wmma_bf16(A0.v, B0.v, acc[0][0].v);
            acc[0][1].v = wmma_bf16(A0.v, B1.v, acc[0][1].v);
            acc[1][0].v = wmma_bf16(A1.v, B0.v, acc[1][0].v);
            acc[1][1].v = wmma_bf16(A1.v, B1.v, acc[1][1].v);
        }
    }
    // bias + relu -> bf16 hidden in LDS
    {
        const float bb0 = b1[nt0 * 16 + lr];
        const float bb1 = b1[nt0 * 16 + 16 + lr];
        #pragma unroll
        for (int mi = 0; mi < 2; ++mi)
            #pragma unroll
            for (int r = 0; r < 8; ++r) {
                int m = (mt0 + mi) * 16 + r + hi * 8;
                hid_sm[m * 136 + nt0 * 16 + lr]      = f2bf(fmaxf(acc[mi][0].f[r] + bb0, 0.f));
                hid_sm[m * 136 + nt0 * 16 + 16 + lr] = f2bf(fmaxf(acc[mi][1].f[r] + bb1, 0.f));
            }
    }
    __syncthreads();

    #pragma unroll
    for (int mi = 0; mi < 2; ++mi)
        #pragma unroll
        for (int ni = 0; ni < 2; ++ni)
            #pragma unroll
            for (int i = 0; i < 8; ++i) acc[mi][ni].f[i] = 0.f;

    // GEMM2: [64x128] @ [128x128]
    {
        const unsigned short* hrow0 = hid_sm + (mt0 * 16 + lr) * 136 + hi * 16;
        const unsigned short* hrow1 = hrow0 + 16 * 136;
        const unsigned short* crow0 = W2t + (nt0 * 16 + lr) * 128 + hi * 16;
        const unsigned short* crow1 = crow0 + 16 * 128;
        #pragma unroll
        for (int kc = 0; kc < 4; ++kc) {
            Frag A0, A1, B0, B1;
            A0.q[0] = *(const uint4*)(hrow0 + kc * 32);
            A0.q[1] = *(const uint4*)(hrow0 + kc * 32 + 8);
            A1.q[0] = *(const uint4*)(hrow1 + kc * 32);
            A1.q[1] = *(const uint4*)(hrow1 + kc * 32 + 8);
            B0.q[0] = *(const uint4*)(crow0 + kc * 32);
            B0.q[1] = *(const uint4*)(crow0 + kc * 32 + 8);
            B1.q[0] = *(const uint4*)(crow1 + kc * 32);
            B1.q[1] = *(const uint4*)(crow1 + kc * 32 + 8);
            acc[0][0].v = wmma_bf16(A0.v, B0.v, acc[0][0].v);
            acc[0][1].v = wmma_bf16(A0.v, B1.v, acc[0][1].v);
            acc[1][0].v = wmma_bf16(A1.v, B0.v, acc[1][0].v);
            acc[1][1].v = wmma_bf16(A1.v, B1.v, acc[1][1].v);
        }
    }

    // Epilogue straight from fragments: e += m (bf16), atomic scatter m -> agg.
    // Fragment VGPR r maps to rows (tile*16 + r + 8*hi), cols nt*16 + lr.
    {
        const float cb0 = b2[nt0 * 16 + lr];
        const float cb1 = b2[nt0 * 16 + 16 + lr];
        const int col0 = nt0 * 16 + lr;
        const int col1 = col0 + 16;
        #pragma unroll
        for (int mi = 0; mi < 2; ++mi) {
            #pragma unroll
            for (int r = 0; r < 8; ++r) {
                int row = (mt0 + mi) * 16 + r + hi * 8;
                long recv = ridx[row];
                float m0 = acc[mi][0].f[r] + cb0;
                float m1 = acc[mi][1].f[r] + cb1;
                float e0 = bf2f(a_sm[row * 392 + col0]) + m0;
                float e1 = bf2f(a_sm[row * 392 + col1]) + m1;
                e_bf[(ebase + row) * LAT + col0] = f2bf(e0);
                e_bf[(ebase + row) * LAT + col1] = f2bf(e1);
                __hip_atomic_fetch_add(agg + recv * LAT + col0, m0,
                                       __ATOMIC_RELAXED, __HIP_MEMORY_SCOPE_AGENT);
                __hip_atomic_fetch_add(agg + recv * LAT + col1, m1,
                                       __ATOMIC_RELAXED, __HIP_MEMORY_SCOPE_AGENT);
            }
        }
    }
}

// ---------------------------------------------------------------------------
// Node update kernel (WMMA): 32 nodes per 256-thread block.
//   dh = relu([h|agg] @ U1 + b1) @ U2 + b2 ; h += dh (f32 + bf16 shadow)
// ---------------------------------------------------------------------------
__global__ __launch_bounds__(256) void node_upd_kernel(
    float* __restrict__ h_f32,
    unsigned short* __restrict__ h_bf,
    const float* __restrict__ agg,
    const unsigned short* __restrict__ W1t,   // [128][256] bf16
    const float* __restrict__ b1,
    const unsigned short* __restrict__ W2t,   // [128][128] bf16
    const float* __restrict__ b2)
{
    __shared__ __align__(16) unsigned short a_sm[32 * 264];   // 32 x 256 (+pad)
    __shared__ __align__(16) unsigned short hid_sm[32 * 136];
    __shared__ __align__(16) float d_sm[32 * 132];

    const int tid = threadIdx.x;
    const long nbase = (long)blockIdx.x * 32;

    // A cols 0..127 = h (bf16 copy)
    for (int c = tid; c < 32 * 16; c += 256) {
        int row = c >> 4, j = c & 15;
        long node = nbase + row;
        uint4 v = make_uint4(0u, 0u, 0u, 0u);
        if (node < NNODES) v = *(((const uint4*)(h_bf + node * LAT)) + j);
        *(uint4*)(a_sm + row * 264 + j * 8) = v;
    }
    // A cols 128..255 = bf16(agg)
    for (int c = tid; c < 32 * 16; c += 256) {
        int row = c >> 4, j = c & 15;
        long node = nbase + row;
        union { unsigned short u[8]; uint4 q; } t;
        if (node < NNODES) {
            const float* ap = agg + node * LAT + j * 8;
            #pragma unroll
            for (int i = 0; i < 8; ++i) t.u[i] = f2bf(ap[i]);
        } else {
            #pragma unroll
            for (int i = 0; i < 8; ++i) t.u[i] = 0;
        }
        *(uint4*)(a_sm + row * 264 + 128 + j * 8) = t.q;
    }
    __syncthreads();

    const int wave = tid >> 5, lane = tid & 31;
    const int mt  = wave >> 2;
    const int nt0 = (wave & 3) * 2;
    const int lr  = lane & 15;
    const int hi  = lane >> 4;

    Acc acc0, acc1;
    #pragma unroll
    for (int i = 0; i < 8; ++i) { acc0.f[i] = 0.f; acc1.f[i] = 0.f; }

    // GEMM1: [32x256] @ [256x128]
    {
        const unsigned short* arow  = a_sm + (mt * 16 + lr) * 264 + hi * 16;
        const unsigned short* brow0 = W1t + (nt0 * 16 + lr) * 256 + hi * 16;
        const unsigned short* brow1 = brow0 + 16 * 256;
        #pragma unroll
        for (int kc = 0; kc < 8; ++kc) {
            Frag A, B0, B1;
            A.q[0]  = *(const uint4*)(arow + kc * 32);
            A.q[1]  = *(const uint4*)(arow + kc * 32 + 8);
            B0.q[0] = *(const uint4*)(brow0 + kc * 32);
            B0.q[1] = *(const uint4*)(brow0 + kc * 32 + 8);
            B1.q[0] = *(const uint4*)(brow1 + kc * 32);
            B1.q[1] = *(const uint4*)(brow1 + kc * 32 + 8);
            acc0.v = wmma_bf16(A.v, B0.v, acc0.v);
            acc1.v = wmma_bf16(A.v, B1.v, acc1.v);
        }
    }
    {
        const float bb0 = b1[nt0 * 16 + lr];
        const float bb1 = b1[nt0 * 16 + 16 + lr];
        #pragma unroll
        for (int r = 0; r < 8; ++r) {
            int m = mt * 16 + r + hi * 8;
            hid_sm[m * 136 + nt0 * 16 + lr]      = f2bf(fmaxf(acc0.f[r] + bb0, 0.f));
            hid_sm[m * 136 + nt0 * 16 + 16 + lr] = f2bf(fmaxf(acc1.f[r] + bb1, 0.f));
        }
    }
    __syncthreads();

    #pragma unroll
    for (int i = 0; i < 8; ++i) { acc0.f[i] = 0.f; acc1.f[i] = 0.f; }

    // GEMM2: [32x128] @ [128x128]
    {
        const unsigned short* hrow  = hid_sm + (mt * 16 + lr) * 136 + hi * 16;
        const unsigned short* brow0 = W2t + (nt0 * 16 + lr) * 128 + hi * 16;
        const unsigned short* brow1 = brow0 + 16 * 128;
        #pragma unroll
        for (int kc = 0; kc < 4; ++kc) {
            Frag A, B0, B1;
            A.q[0]  = *(const uint4*)(hrow + kc * 32);
            A.q[1]  = *(const uint4*)(hrow + kc * 32 + 8);
            B0.q[0] = *(const uint4*)(brow0 + kc * 32);
            B0.q[1] = *(const uint4*)(brow0 + kc * 32 + 8);
            B1.q[0] = *(const uint4*)(brow1 + kc * 32);
            B1.q[1] = *(const uint4*)(brow1 + kc * 32 + 8);
            acc0.v = wmma_bf16(A.v, B0.v, acc0.v);
            acc1.v = wmma_bf16(A.v, B1.v, acc1.v);
        }
    }
    {
        const float cb0 = b2[nt0 * 16 + lr];
        const float cb1 = b2[nt0 * 16 + 16 + lr];
        #pragma unroll
        for (int r = 0; r < 8; ++r) {
            int m = mt * 16 + r + hi * 8;
            d_sm[m * 132 + nt0 * 16 + lr]      = acc0.f[r] + cb0;
            d_sm[m * 132 + nt0 * 16 + 16 + lr] = acc1.f[r] + cb1;
        }
    }
    __syncthreads();

    // Residual: h += dh
    for (int i = tid; i < 32 * 128; i += 256) {
        int row = i >> 7, col = i & 127;
        long node = nbase + row;
        if (node < NNODES) {
            float v = h_f32[node * LAT + col] + d_sm[row * 132 + col];
            h_f32[node * LAT + col] = v;
            h_bf[node * LAT + col]  = f2bf(v);
        }
    }
}

// ---------------------------------------------------------------------------
// Decoder: out = MLP2(h)[N,2] * out_std + out_mean
// ---------------------------------------------------------------------------
__global__ __launch_bounds__(128) void decoder_kernel(
    const float* __restrict__ h_f32,
    const float* __restrict__ W1, const float* __restrict__ b1,
    const float* __restrict__ W2, const float* __restrict__ b2,
    const float* __restrict__ out_mean, const float* __restrict__ out_std,
    float* __restrict__ out)
{
    __shared__ float x_sm[8][129];
    __shared__ float hid_sm[8][129];
    const long nbase = (long)blockIdx.x * 8;
    const int tid = threadIdx.x;
    #pragma unroll
    for (int e = 0; e < 8; ++e) x_sm[e][tid] = h_f32[(nbase + e) * LAT + tid];
    __syncthreads();
    float acc[8];
    #pragma unroll
    for (int e = 0; e < 8; ++e) acc[e] = b1[tid];
    for (int k = 0; k < LAT; ++k) {
        float w = W1[k * LAT + tid];
        #pragma unroll
        for (int e = 0; e < 8; ++e) acc[e] += x_sm[e][k] * w;
    }
    #pragma unroll
    for (int e = 0; e < 8; ++e) hid_sm[e][tid] = fmaxf(acc[e], 0.f);
    __syncthreads();
    if (tid < 16) {
        int e = tid >> 1, j = tid & 1;
        float s = b2[j];
        for (int k = 0; k < LAT; ++k) s += hid_sm[e][k] * W2[k * 2 + j];
        out[(nbase + e) * 2 + j] = s * out_std[j] + out_mean[j];
    }
}

// ---------------------------------------------------------------------------
extern "C" void kernel_launch(void* const* d_in, const int* in_sizes, int n_in,
                              void* d_out, int out_size, void* d_ws, size_t ws_size,
                              hipStream_t stream)
{
    (void)in_sizes; (void)n_in; (void)out_size; (void)ws_size;
    const float* node_in   = (const float*)d_in[0];
    const float* edge_in   = (const float*)d_in[1];
    const int*   senders   = (const int*)d_in[2];
    const int*   receivers = (const int*)d_in[3];
    const float* node_mean = (const float*)d_in[4];
    const float* node_std  = (const float*)d_in[5];
    const float* edge_mean = (const float*)d_in[6];
    const float* edge_std  = (const float*)d_in[7];
    const float* out_mean  = (const float*)d_in[8];
    const float* out_std   = (const float*)d_in[9];
    const float* ne_W1 = (const float*)d_in[10];
    const float* ne_b1 = (const float*)d_in[11];
    const float* ne_W2 = (const float*)d_in[12];
    const float* ne_b2 = (const float*)d_in[13];
    const float* ee_W1 = (const float*)d_in[14];
    const float* ee_b1 = (const float*)d_in[15];
    const float* ee_W2 = (const float*)d_in[16];
    const float* ee_b2 = (const float*)d_in[17];
    const float* m_W1  = (const float*)d_in[18];
    const float* m_b1  = (const float*)d_in[19];
    const float* m_W2  = (const float*)d_in[20];
    const float* m_b2  = (const float*)d_in[21];
    const float* u_W1  = (const float*)d_in[22];
    const float* u_b1  = (const float*)d_in[23];
    const float* u_W2  = (const float*)d_in[24];
    const float* u_b2  = (const float*)d_in[25];
    const float* dW1   = (const float*)d_in[26];
    const float* db1   = (const float*)d_in[27];
    const float* dW2   = (const float*)d_in[28];
    const float* db2   = (const float*)d_in[29];
    float* out = (float*)d_out;

    char* ws = (char*)d_ws;
    size_t off = 0;
    auto carve = [&](size_t bytes) -> void* {
        void* p = ws + off;
        off += (bytes + 255) & ~(size_t)255;
        return p;
    };
    float*          h_f32 = (float*)carve((size_t)NNODES * LAT * 4);
    unsigned short* h_bf  = (unsigned short*)carve((size_t)NNODES * LAT * 2);
    unsigned short* e_bf  = (unsigned short*)carve((size_t)NEDGES * LAT * 2);
    float*          agg   = (float*)carve((size_t)NNODES * LAT * 4);
    unsigned short* mW1t  = (unsigned short*)carve((size_t)NSTEPS * LAT * 384 * 2);
    unsigned short* mW2t  = (unsigned short*)carve((size_t)NSTEPS * LAT * LAT * 2);
    unsigned short* uW1t  = (unsigned short*)carve((size_t)NSTEPS * LAT * 256 * 2);
    unsigned short* uW2t  = (unsigned short*)carve((size_t)NSTEPS * LAT * LAT * 2);

    // Weight prep (bf16, N-major for WMMA B fragments)
    transpose_bf16_kernel<<<1024, 256, 0, stream>>>(m_W1, mW1t, NSTEPS, 384, LAT);
    transpose_bf16_kernel<<<1024, 256, 0, stream>>>(m_W2, mW2t, NSTEPS, LAT, LAT);
    transpose_bf16_kernel<<<1024, 256, 0, stream>>>(u_W1, uW1t, NSTEPS, 256, LAT);
    transpose_bf16_kernel<<<1024, 256, 0, stream>>>(u_W2, uW2t, NSTEPS, LAT, LAT);

    // Encoders
    node_enc_kernel<<<NNODES / 8, 128, 0, stream>>>(node_in, node_mean, node_std,
        ne_W1, ne_b1, ne_W2, ne_b2, h_f32, h_bf);
    edge_enc_kernel<<<NEDGES / 16, 128, 0, stream>>>(edge_in, edge_mean, edge_std,
        ee_W1, ee_b1, ee_W2, ee_b2, e_bf);

    // Message-passing steps
    for (int s = 0; s < NSTEPS; ++s) {
        hipMemsetAsync(agg, 0, (size_t)NNODES * LAT * 4, stream);
        edge_msg_kernel<<<NEDGES / 64, 256, EMSG_LDS_BYTES, stream>>>(
            e_bf, h_bf, senders, receivers,
            mW1t + (size_t)s * LAT * 384, m_b1 + (size_t)s * LAT,
            mW2t + (size_t)s * LAT * LAT, m_b2 + (size_t)s * LAT,
            agg);
        node_upd_kernel<<<(NNODES + 31) / 32, 256, 0, stream>>>(
            h_f32, h_bf, agg,
            uW1t + (size_t)s * LAT * 256, u_b1 + (size_t)s * LAT,
            uW2t + (size_t)s * LAT * LAT, u_b2 + (size_t)s * LAT);
    }

    // Decoder
    decoder_kernel<<<NNODES / 8, 128, 0, stream>>>(h_f32, dW1, db1, dW2, db2,
        out_mean, out_std, out);
}
